// AttentionLayer_81587198755136
// MI455X (gfx1250) — matrix operand order repaired and checked by reading, not compiled, not run on hardware
//
#include <hip/hip_runtime.h>

// ---------------------------------------------------------------------------
// Problem constants (match reference)
// ---------------------------------------------------------------------------
#define BATCH   2
#define TSEQ    2048
#define DMODEL  2048
#define NH      16
#define HDQ     64          // sem & geo head dim
#define VHD     128         // value head dim
#define SEMD    1024        // NH * HDQ (sem == geo width)
#define VDTOT   2048        // NH * VHD
#define BT_TOT  (BATCH * TSEQ)   // 4096

typedef __bf16 bf16;
typedef __attribute__((ext_vector_type(16))) __bf16 v16bf;
typedef __attribute__((ext_vector_type(8)))  __bf16 v8bf;
typedef __attribute__((ext_vector_type(8)))  float  v8f;

#define WMMA_BF16(A, B, C) \
  __builtin_amdgcn_wmma_f32_16x16x32_bf16(false, (A), false, (B), (short)0, (C), false, false)

// Two contiguous 16B loads -> one 16-element bf16 fragment (8 VGPRs).
__device__ __forceinline__ v16bf frag_cat(const bf16* p0, const bf16* p1) {
  v8bf lo = *(const v8bf*)p0;
  v8bf hi = *(const v8bf*)p1;
  return __builtin_shufflevector(lo, hi, 0,1,2,3,4,5,6,7,8,9,10,11,12,13,14,15);
}

// LDS byte address (ISA: LDS aperture address = addr[31:0]).
__device__ __forceinline__ unsigned lds_addr(const void* p) {
  return (unsigned)(uintptr_t)p;
}

// Async DMA: 16 bytes global -> LDS, tracked by ASYNCcnt (CDNA5).
__device__ __forceinline__ void async_copy_b128(unsigned lds, const void* gptr) {
  asm volatile("global_load_async_to_lds_b128 %0, %1, off"
               :: "v"(lds), "v"((unsigned long long)(uintptr_t)gptr)
               : "memory");
}

// ---------------------------------------------------------------------------
// Tiled GEMM: C[M,N] = A[M,K] * Wt[N,K]^T (both bf16), f32 accumulate.
// Block = 256 threads (8 waves), tile 128x128, BK = 32.
// Double-buffered LDS staging via async global->LDS DMA (ASYNCcnt).
// ---------------------------------------------------------------------------
template <bool OUT_BF16>
__global__ __launch_bounds__(256) void gemm_tile(const bf16* __restrict__ A,
                                                 const bf16* __restrict__ Wt,
                                                 void* __restrict__ Cp,
                                                 int N, int K) {
  __shared__ bf16 As[2][128][40];   // [buf][m][k], +8 pad
  __shared__ bf16 Bs[2][128][40];   // [buf][n][k], +8 pad

  const int tid  = threadIdx.x;
  const int lane = tid & 31;
  const int wid  = tid >> 5;
  const int ln   = lane & 15;
  const int half = lane >> 4;
  const int wr   = wid >> 2;     // 0..1  (64-row strip)
  const int wc   = wid & 3;      // 0..3  (32-col strip)
  const int m0   = blockIdx.y * 128;
  const int n0   = blockIdx.x * 128;
  const int koA  = half * 16;    // A-frag K offsets: {0..15} / {16..31}
  const int koB  = half * 16;    // B-frag K offsets: {0..15} / {16..31}

  // 4 async b128 copies per thread per k-tile (2 for A, 2 for B)
  auto issue_tile = [&](int buf, int k0) {
#pragma unroll
    for (int it = 0; it < 2; ++it) {
      int row = (tid >> 2) + it * 64;
      int col = (tid & 3) * 8;
      async_copy_b128(lds_addr(&As[buf][row][col]),
                      A + (size_t)(m0 + row) * K + k0 + col);
      async_copy_b128(lds_addr(&Bs[buf][row][col]),
                      Wt + (size_t)(n0 + row) * K + k0 + col);
    }
  };

  v8f acc[4][2];
#pragma unroll
  for (int mi = 0; mi < 4; ++mi)
#pragma unroll
    for (int ni = 0; ni < 2; ++ni) acc[mi][ni] = (v8f){0,0,0,0,0,0,0,0};

  const int KT = K / 32;
  issue_tile(0, 0);
  for (int kb = 0; kb < KT; ++kb) {
    const int cur = kb & 1;
    if (kb + 1 < KT) {
      issue_tile(cur ^ 1, (kb + 1) * 32);        // prefetch next tile
      asm volatile("s_wait_asynccnt 0x4" ::: "memory");  // cur's 4 copies done
    } else {
      asm volatile("s_wait_asynccnt 0x0" ::: "memory");
    }
    __syncthreads();                              // all waves' copies visible

    v16bf af[4], bfr[2];
#pragma unroll
    for (int mi = 0; mi < 4; ++mi) {
      int row = wr * 64 + mi * 16 + ln;
      af[mi] = frag_cat(&As[cur][row][koA], &As[cur][row][koA + 8]);
    }
#pragma unroll
    for (int ni = 0; ni < 2; ++ni) {
      int col = wc * 32 + ni * 16 + ln;
      bfr[ni] = frag_cat(&Bs[cur][col][koB], &Bs[cur][col][koB + 8]);
    }
#pragma unroll
    for (int mi = 0; mi < 4; ++mi)
#pragma unroll
      for (int ni = 0; ni < 2; ++ni)
        acc[mi][ni] = WMMA_BF16(af[mi], bfr[ni], acc[mi][ni]);

    __syncthreads();                              // done reading cur buffer
  }

  // ---- epilogue: C layout row = r + 8*half (per lane), col = lane%16 ----
#pragma unroll
  for (int mi = 0; mi < 4; ++mi)
#pragma unroll
    for (int ni = 0; ni < 2; ++ni)
#pragma unroll
      for (int r = 0; r < 8; ++r) {
        int row = m0 + wr * 64 + mi * 16 + r + half * 8;
        int col = n0 + wc * 32 + ni * 16 + ln;
        float v = acc[mi][ni][r];
        if constexpr (OUT_BF16)
          ((bf16*)Cp)[(size_t)row * N + col] = (bf16)v;
        else
          ((float*)Cp)[(size_t)row * N + col] = v;
      }
}

// ---------------------------------------------------------------------------
// One-shot f32 -> bf16 cast (8 elems / thread).
// ---------------------------------------------------------------------------
__global__ __launch_bounds__(256) void cast_bf16_kernel(const float* __restrict__ src,
                                                        bf16* __restrict__ dst) {
  size_t i = ((size_t)blockIdx.x * 256 + threadIdx.x) * 8;
  float4 a = *(const float4*)(src + i);
  float4 b = *(const float4*)(src + i + 4);
  v8bf v = {(bf16)a.x, (bf16)a.y, (bf16)a.z, (bf16)a.w,
            (bf16)b.x, (bf16)b.y, (bf16)b.z, (bf16)b.w};
  *(v8bf*)(dst + i) = v;
}

// ---------------------------------------------------------------------------
// One-shot weight transpose + cast: W[K=2048][N] f32 -> Wt[N][K] bf16.
// Output-coalesced: each thread writes 8 consecutive k as one b128 store.
// ---------------------------------------------------------------------------
__global__ __launch_bounds__(256) void wtrans_kernel(const float* __restrict__ W,
                                                     bf16* __restrict__ Wt, int N) {
  int gid = blockIdx.x * 256 + threadIdx.x;        // N * (2048/8)
  int kc  = (gid & 255) * 8;
  int n   = gid >> 8;
  v8bf v;
#pragma unroll
  for (int j = 0; j < 8; ++j) v[j] = (bf16)W[(size_t)(kc + j) * N + n];
  *(v8bf*)(Wt + (size_t)n * DMODEL + kc) = v;
}

// ---------------------------------------------------------------------------
// dyn = x @ gate_w   (4096 x 2048 @ 2048 x 16) — tiny, float4 VALU.
// ---------------------------------------------------------------------------
__global__ __launch_bounds__(256) void dyn_kernel(const float* __restrict__ x,
                                                  const float* __restrict__ gw,
                                                  float* __restrict__ dynb) {
  int gid = blockIdx.x * 256 + threadIdx.x;  // BT_TOT * NH threads
  int h   = gid & (NH - 1);
  int row = gid >> 4;
  const float* xr = x + (size_t)row * DMODEL;
  float s = 0.0f;
  for (int k = 0; k < DMODEL; k += 4) {
    float4 xv = *(const float4*)(xr + k);
    s += xv.x * gw[(size_t)(k + 0) * NH + h];
    s += xv.y * gw[(size_t)(k + 1) * NH + h];
    s += xv.z * gw[(size_t)(k + 2) * NH + h];
    s += xv.w * gw[(size_t)(k + 3) * NH + h];
  }
  dynb[gid] = s;
}

// ---------------------------------------------------------------------------
// In-place RoPE on bf16 geo Q/K buffers. One thread per (bt, h, i<32) pair.
// ---------------------------------------------------------------------------
__global__ __launch_bounds__(256) void rope_kernel(bf16* __restrict__ qg,
                                                   bf16* __restrict__ kg) {
  int gid = blockIdx.x * 256 + threadIdx.x;      // BT_TOT * NH * 32
  int i  = gid & 31;
  int h  = (gid >> 5) & (NH - 1);
  int bt = gid >> 9;
  int t  = bt & (TSEQ - 1);
  float inv = __expf(-(float)i * (9.210340371976184f / 32.0f));  // 10000^(-i/32)
  float ang = (float)t * inv;
  float sn, cs;
  __sincosf(ang, &sn, &cs);
  size_t base = (size_t)bt * SEMD + (size_t)h * HDQ + i;
  float a1 = (float)qg[base], a2 = (float)qg[base + 32];
  qg[base]      = (bf16)(a1 * cs - a2 * sn);
  qg[base + 32] = (bf16)(a2 * cs + a1 * sn);
  float b1 = (float)kg[base], b2 = (float)kg[base + 32];
  kg[base]      = (bf16)(b1 * cs - b2 * sn);
  kg[base + 32] = (bf16)(b2 * cs + b1 * sn);
}

// ---------------------------------------------------------------------------
// One-shot V transpose: vbuf[(b*T+t)][h*VHD+d] -> vT[((b*NH+h)*VHD+d)][T]
// ---------------------------------------------------------------------------
__global__ __launch_bounds__(256) void vtrans_kernel(const bf16* __restrict__ vbuf,
                                                     bf16* __restrict__ vT) {
  int gid  = blockIdx.x * 256 + threadIdx.x;       // BATCH*NH*VHD * (T/8)
  int tcol = (gid & (TSEQ / 8 - 1)) * 8;
  int orow = gid >> 8;                             // 0..4095 = (b*NH+h)*VHD+d
  int b  = orow >> 11;
  int hd = orow & 2047;
  int h  = hd >> 7;
  int d  = hd & 127;
  v8bf v;
#pragma unroll
  for (int j = 0; j < 8; ++j)
    v[j] = vbuf[((size_t)(b * TSEQ) + tcol + j) * VDTOT + (size_t)h * VHD + d];
  *(v8bf*)(vT + (size_t)orow * TSEQ + tcol) = v;
}

// ---------------------------------------------------------------------------
// Flash attention with gated sem/geo mix (barrier-free, per-wave causal loop).
// grid = (T/128, B*H); block = 256 (8 waves), each wave owns 16 query rows.
// ---------------------------------------------------------------------------
__global__ __launch_bounds__(256) void attn_kernel(
    const bf16* __restrict__ qs, const bf16* __restrict__ ks,
    const bf16* __restrict__ qg, const bf16* __restrict__ kg,
    const bf16* __restrict__ vT, const float* __restrict__ dynb,
    const float* __restrict__ gate_logit, bf16* __restrict__ ob) {
  __shared__ bf16 Pb[8][16][40];    // per-wave P staging (C-layout -> A-frag)

  const int tid  = threadIdx.x;
  const int lane = tid & 31;
  const int wid  = tid >> 5;
  const int ln   = lane & 15;
  const int half = lane >> 4;
  const int bh = blockIdx.y;
  const int b  = bh >> 4;
  const int h  = bh & (NH - 1);
  const int q0 = blockIdx.x * 128 + wid * 16;
  const int koA = half * 16;
  const int koB = half * 16;

  const size_t qrow = (size_t)(b * TSEQ) + q0 + ln;
  const bf16* qsp = qs + qrow * SEMD + (size_t)h * HDQ;
  const bf16* qgp = qg + qrow * SEMD + (size_t)h * HDQ;
  v16bf qsf0 = frag_cat(qsp + koA,      qsp + koA + 8);
  v16bf qsf1 = frag_cat(qsp + 32 + koA, qsp + 32 + koA + 8);
  v16bf qgf0 = frag_cat(qgp + koA,      qgp + koA + 8);
  v16bf qgf1 = frag_cat(qgp + 32 + koA, qgp + 32 + koA + 8);

  const float gl = gate_logit[h];
  float gv[8];
#pragma unroll
  for (int r = 0; r < 8; ++r) {
    int t = q0 + r + half * 8;
    float z = gl + dynb[((size_t)(b * TSEQ) + t) * NH + h];
    gv[r] = 1.0f / (1.0f + __expf(-z));
  }

  float m[8], l[8];
  v8f acc[8];
#pragma unroll
  for (int r = 0; r < 8; ++r) { m[r] = -1e30f; l[r] = 0.0f; }
#pragma unroll
  for (int n = 0; n < 8; ++n) acc[n] = (v8f){0,0,0,0,0,0,0,0};

  const bf16* vhead = vT + (size_t)(b * NH + h) * VHD * TSEQ;
  bf16* Pw = &Pb[wid][0][0];

  const int nkt = q0 / 32 + 1;      // wave-private causal key-tile count
  for (int kt = 0; kt < nkt; ++kt) {
    v16bf kf[2][2], gf[2][2];
#pragma unroll
    for (int tile = 0; tile < 2; ++tile) {
      int key = kt * 32 + tile * 16 + ln;
      const bf16* kp = ks + ((size_t)(b * TSEQ) + key) * SEMD + (size_t)h * HDQ;
      const bf16* gp = kg + ((size_t)(b * TSEQ) + key) * SEMD + (size_t)h * HDQ;
      kf[tile][0] = frag_cat(kp + koB,      kp + koB + 8);
      kf[tile][1] = frag_cat(kp + 32 + koB, kp + 32 + koB + 8);
      gf[tile][0] = frag_cat(gp + koB,      gp + koB + 8);
      gf[tile][1] = frag_cat(gp + 32 + koB, gp + 32 + koB + 8);
    }
    v8f ssem[2], sgeo[2];
#pragma unroll
    for (int tile = 0; tile < 2; ++tile) {
      v8f s = (v8f){0,0,0,0,0,0,0,0};
      s = WMMA_BF16(qsf0, kf[tile][0], s);
      s = WMMA_BF16(qsf1, kf[tile][1], s);
      ssem[tile] = s;
      v8f g = (v8f){0,0,0,0,0,0,0,0};
      g = WMMA_BF16(qgf0, gf[tile][0], g);
      g = WMMA_BF16(qgf1, gf[tile][1], g);
      sgeo[tile] = g;
    }

    float sc[8], p0a[8], p1a[8];
#pragma unroll
    for (int r = 0; r < 8; ++r) {
      int t = q0 + r + half * 8;
      float ge = gv[r];
      float s0 = (ge * ssem[0][r] + (1.0f - ge) * sgeo[0][r]) * 0.125f;
      float s1 = (ge * ssem[1][r] + (1.0f - ge) * sgeo[1][r]) * 0.125f;
      int c0 = kt * 32 + ln;
      if (c0 > t)      s0 = -1e9f;
      if (c0 + 16 > t) s1 = -1e9f;
      float mx = fmaxf(s0, s1);
#pragma unroll
      for (int o = 1; o < 16; o <<= 1) mx = fmaxf(mx, __shfl_xor(mx, o, 32));
      float mn    = fmaxf(m[r], mx);
      float scale = __expf(m[r] - mn);
      float p0 = __expf(s0 - mn);
      float p1 = __expf(s1 - mn);
      float rs = p0 + p1;
#pragma unroll
      for (int o = 1; o < 16; o <<= 1) rs += __shfl_xor(rs, o, 32);
      l[r] = l[r] * scale + rs;
      m[r] = mn;
      sc[r]  = scale;
      p0a[r] = p0;
      p1a[r] = p1;
    }
#pragma unroll
    for (int n = 0; n < 8; ++n)
#pragma unroll
      for (int r = 0; r < 8; ++r) acc[n][r] *= sc[r];

    v16bf vf[8];
#pragma unroll
    for (int n = 0; n < 8; ++n) {
      const bf16* vp = vhead + (size_t)(n * 16 + ln) * TSEQ + kt * 32 + koB;
      vf[n] = frag_cat(vp, vp + 8);
    }

#pragma unroll
    for (int r = 0; r < 8; ++r) {
      int row = r + half * 8;
      Pw[row * 40 + ln]      = (bf16)p0a[r];
      Pw[row * 40 + 16 + ln] = (bf16)p1a[r];
    }
    v16bf pf = frag_cat(&Pw[ln * 40 + koA], &Pw[ln * 40 + koA + 8]);

#pragma unroll
    for (int n = 0; n < 8; ++n) acc[n] = WMMA_BF16(pf, vf[n], acc[n]);
  }

  float rl[8];
#pragma unroll
  for (int r = 0; r < 8; ++r) rl[r] = 1.0f / l[r];
#pragma unroll
  for (int n = 0; n < 8; ++n)
#pragma unroll
    for (int r = 0; r < 8; ++r) {
      int t = q0 + r + half * 8;
      ob[((size_t)(b * TSEQ) + t) * VDTOT + (size_t)h * VHD + n * 16 + ln] =
          (bf16)(acc[n][r] * rl[r]);
    }
}

// ---------------------------------------------------------------------------
// Host launcher
// ---------------------------------------------------------------------------
extern "C" void kernel_launch(void* const* d_in, const int* in_sizes, int n_in,
                              void* d_out, int out_size, void* d_ws, size_t ws_size,
                              hipStream_t stream) {
  (void)in_sizes; (void)n_in; (void)out_size; (void)ws_size;

  const float* x        = (const float*)d_in[0];
  const float* w_q_sem  = (const float*)d_in[1];
  const float* w_k_sem  = (const float*)d_in[2];
  const float* w_q_geo  = (const float*)d_in[3];
  const float* w_k_geo  = (const float*)d_in[4];
  const float* w_v      = (const float*)d_in[5];
  const float* w_out    = (const float*)d_in[6];
  const float* glogit   = (const float*)d_in[7];
  const float* gate_w   = (const float*)d_in[8];
  float* y = (float*)d_out;

  // workspace carve-up
  char* ws = (char*)d_ws;
  bf16* xb    = (bf16*)ws;  ws += (size_t)BT_TOT * DMODEL * sizeof(bf16);
  bf16* wqs_t = (bf16*)ws;  ws += (size_t)SEMD  * DMODEL * sizeof(bf16);
  bf16* wks_t = (bf16*)ws;  ws += (size_t)SEMD  * DMODEL * sizeof(bf16);
  bf16* wqg_t = (bf16*)ws;  ws += (size_t)SEMD  * DMODEL * sizeof(bf16);
  bf16* wkg_t = (bf16*)ws;  ws += (size_t)SEMD  * DMODEL * sizeof(bf16);
  bf16* wv_t  = (bf16*)ws;  ws += (size_t)VDTOT * DMODEL * sizeof(bf16);
  bf16* wo_t  = (bf16*)ws;  ws += (size_t)DMODEL * VDTOT * sizeof(bf16);
  bf16* qsb   = (bf16*)ws;  ws += (size_t)BT_TOT * SEMD  * sizeof(bf16);
  bf16* ksb   = (bf16*)ws;  ws += (size_t)BT_TOT * SEMD  * sizeof(bf16);
  bf16* qgb   = (bf16*)ws;  ws += (size_t)BT_TOT * SEMD  * sizeof(bf16);
  bf16* kgb   = (bf16*)ws;  ws += (size_t)BT_TOT * SEMD  * sizeof(bf16);
  bf16* vbuf  = (bf16*)ws;  ws += (size_t)BT_TOT * VDTOT * sizeof(bf16);
  bf16* vTb   = (bf16*)ws;  ws += (size_t)BT_TOT * VDTOT * sizeof(bf16);
  bf16* attn  = (bf16*)ws;  ws += (size_t)BT_TOT * VDTOT * sizeof(bf16);
  float* dynb = (float*)ws; ws += (size_t)BT_TOT * NH    * sizeof(float);

  dim3 blk(256);

  // 0) one-shot precision/layout prep (bandwidth-trivial)
  cast_bf16_kernel<<<(BT_TOT * DMODEL / 8) / 256, blk, 0, stream>>>(x, xb);
  wtrans_kernel<<<SEMD,  blk, 0, stream>>>(w_q_sem, wqs_t, SEMD);
  wtrans_kernel<<<SEMD,  blk, 0, stream>>>(w_k_sem, wks_t, SEMD);
  wtrans_kernel<<<SEMD,  blk, 0, stream>>>(w_q_geo, wqg_t, SEMD);
  wtrans_kernel<<<SEMD,  blk, 0, stream>>>(w_k_geo, wkg_t, SEMD);
  wtrans_kernel<<<VDTOT, blk, 0, stream>>>(w_v,   wv_t, VDTOT);
  wtrans_kernel<<<VDTOT, blk, 0, stream>>>(w_out, wo_t, DMODEL);

  // 1) projections (async-DMA staged bf16 GEMMs)
  dim3 gq(SEMD / 128, BT_TOT / 128);
  gemm_tile<true><<<gq, blk, 0, stream>>>(xb, wqs_t, qsb, SEMD, DMODEL);
  gemm_tile<true><<<gq, blk, 0, stream>>>(xb, wks_t, ksb, SEMD, DMODEL);
  gemm_tile<true><<<gq, blk, 0, stream>>>(xb, wqg_t, qgb, SEMD, DMODEL);
  gemm_tile<true><<<gq, blk, 0, stream>>>(xb, wkg_t, kgb, SEMD, DMODEL);
  dim3 gvv(VDTOT / 128, BT_TOT / 128);
  gemm_tile<true><<<gvv, blk, 0, stream>>>(xb, wv_t, vbuf, VDTOT, DMODEL);

  // 2) dynamic gate logits
  dyn_kernel<<<(BT_TOT * NH) / 256, blk, 0, stream>>>(x, gate_w, dynb);

  // 3) RoPE on geo q/k
  rope_kernel<<<(BT_TOT * NH * 32) / 256, blk, 0, stream>>>(qgb, kgb);

  // 4) one-shot V transpose to [b][h][dim][T]
  vtrans_kernel<<<(BATCH * NH * VHD * (TSEQ / 8)) / 256, blk, 0, stream>>>(vbuf, vTb);

  // 5) gated flash attention (barrier-free, per-wave causal loop)
  dim3 ga(TSEQ / 128, BATCH * NH);
  attn_kernel<<<ga, blk, 0, stream>>>(qsb, ksb, qgb, kgb, vTb, dynb, glogit, attn);

  // 6) output projection (bf16 A -> f32 C into d_out)
  dim3 go(DMODEL / 128, BT_TOT / 128);
  gemm_tile<false><<<go, blk, 0, stream>>>(attn, wo_t, y, DMODEL, VDTOT);
}
